// SafeGATWrapper_51805895524973
// MI455X (gfx1250) — matrix operand order repaired and checked by reading, not compiled, not run on hardware
//
#include <hip/hip_runtime.h>
#include <hip/hip_bf16.h>

// GAT layer for MI455X (gfx1250, wave32).
// init(out=bias, amax=0, denom=0) -> pack W (f16, WMMA-B lane layout)
//  -> WMMA GEMM h = x@W (B staged in LDS, f16 in, f32 acc, h stored f16)
//  -> node scores -> edge segment-max (uint-encoded atomicMax)
//  -> edge segment-sum of exp -> weighted scatter-add (1 wave/edge, 256B gathers).

typedef __attribute__((ext_vector_type(16))) _Float16 v16h;
typedef __attribute__((ext_vector_type(8)))  _Float16 v8h;
typedef __attribute__((ext_vector_type(4)))  _Float16 v4h;
typedef __attribute__((ext_vector_type(8)))  float    v8f;

#define IN_F      128
#define OUT_F     32
#define HEADS     4
#define NEG_SLOPE 0.2f
#define WP_HALVES (4 * 8 * 32 * 16)   // 16384 halves = 32 KB packed W

// ---- monotonic float <-> uint encoding for atomicMax over floats ----
__device__ __forceinline__ unsigned enc_f32(float f) {
  unsigned u = __float_as_uint(f);
  return (u & 0x80000000u) ? ~u : (u | 0x80000000u);
}
__device__ __forceinline__ float dec_f32(unsigned k) {
  unsigned u = (k & 0x80000000u) ? (k ^ 0x80000000u) : ~k;
  return __uint_as_float(u);
}

__device__ __forceinline__ void edge_nodes(const int* __restrict__ ei, int e, int E_,
                                           int& s, int& d) {
  if (e < E_) { s = ei[e]; d = ei[E_ + e]; }
  else        { s = e - E_; d = s; }          // appended self-loops
}

// out[n,128] = bias[j]; amax=0 (== -inf sentinel); denom=0
__global__ void k_init(float* __restrict__ out, unsigned* __restrict__ amax,
                       float* __restrict__ denom, const float* __restrict__ bias, int n) {
  int idx = blockIdx.x * blockDim.x + threadIdx.x;
  if (idx < n * IN_F) out[idx] = bias[idx & (IN_F - 1)];
  if (idx < n * HEADS) { amax[idx] = 0u; denom[idx] = 0.0f; }
}

// Pack W[128,128] fp32 -> f16 in per-lane WMMA B layout:
//   chunk ((kt*8 + nt)*32 + lane), 16 contiguous halves per lane:
//   element i -> W[kt*32 + (lane>>4)*16 + i][nt*16 + (lane&15)]
__global__ void k_pack_w(const float* __restrict__ W, _Float16* __restrict__ Wp) {
  int idx = blockIdx.x * blockDim.x + threadIdx.x;   // 0 .. 16383
  if (idx >= WP_HALVES) return;
  int i    = idx & 15;
  int lane = (idx >> 4) & 31;
  int nt   = (idx >> 9) & 7;
  int kt   = idx >> 12;
  int k = kt * 32 + (lane >> 4) * 16 + i;
  int c = nt * 16 + (lane & 15);
  Wp[idx] = (_Float16)W[k * IN_F + c];
}

// h = x @ W via v_wmma_f32_16x16x32_f16.
// 4 waves/block, one wave per 16-row tile; packed W staged once in LDS (32 KB)
// so the inner loop feeds B from ds_load_b128 instead of L2.
__global__ void __launch_bounds__(128) k_gemm(const float* __restrict__ x,
                                              const _Float16* __restrict__ Wp,
                                              _Float16* __restrict__ h,
                                              int n, int ntiles) {
  __shared__ _Float16 lwp[WP_HALVES];

  // cooperative stage: 2048 x 16B chunks over 128 threads
  {
    const uint4* src = (const uint4*)Wp;
    uint4*       dst = (uint4*)lwp;
    #pragma unroll
    for (int i = 0; i < (WP_HALVES * 2 / 16) / 128; ++i)
      dst[threadIdx.x + i * 128] = src[threadIdx.x + i * 128];
  }
  __syncthreads();

  int wave = (blockIdx.x * blockDim.x + threadIdx.x) >> 5;   // uniform per wave
  if (wave >= ntiles) return;                                 // whole-wave exit
  int lane  = threadIdx.x & 31;
  int r0    = wave * 16;
  int lrow  = r0 + (lane & 15);
  if (lrow >= n) lrow = n - 1;        // clamp loads; keeps EXEC all-ones
  int khalf = lane >> 4;              // A-operand K-half select

  v8f acc[8] = {};                    // 8 N-tiles of 16x16 f32 accumulators

  #pragma unroll
  for (int kt = 0; kt < 4; ++kt) {
    // A (16x32 f16): lane holds K = kt*32 + khalf*16 + {0..15}
    const float* xr = x + (size_t)lrow * IN_F + kt * 32 + khalf * 16;
    v16h a;
    #pragma unroll
    for (int i = 0; i < 8; ++i) a[i]     = (_Float16)xr[i];
    #pragma unroll
    for (int i = 0; i < 8; ++i) a[8 + i] = (_Float16)xr[8 + i];

    #pragma unroll
    for (int nt = 0; nt < 8; ++nt) {
      v16h b = ((const v16h*)lwp)[(kt * 8 + nt) * 32 + lane];  // LDS, 32B/lane
      acc[nt] = __builtin_amdgcn_wmma_f32_16x16x32_f16(
          /*neg_a=*/false, a, /*neg_b=*/false, b,
          /*c_mod=*/(short)0, acc[nt], /*reuse_a=*/false, /*reuse_b=*/false);
    }
  }

  // D layout: lane -> col (lane&15), VGPR j -> row j + 8*khalf
  int colb = lane & 15;
  #pragma unroll
  for (int nt = 0; nt < 8; ++nt) {
    #pragma unroll
    for (int j = 0; j < 8; ++j) {
      int row = r0 + j + 8 * khalf;
      if (row < n) h[(size_t)row * IN_F + nt * 16 + colb] = (_Float16)acc[nt][j];
    }
  }
}

// a_src[n,h] = <h[n,h,:], att_src[h,:]>, a_dst likewise (b128 reads of h)
__global__ void k_scores(const _Float16* __restrict__ h,
                         const float* __restrict__ att_src, const float* __restrict__ att_dst,
                         float* __restrict__ asrc, float* __restrict__ adst, int n) {
  int idx = blockIdx.x * blockDim.x + threadIdx.x;
  if (idx >= n * HEADS) return;
  int node = idx >> 2, hd = idx & 3;
  const v8h* hp = (const v8h*)(h + (size_t)node * IN_F + hd * OUT_F);  // 16B aligned
  float s = 0.f, d = 0.f;
  #pragma unroll
  for (int blk = 0; blk < 4; ++blk) {
    v8h v = hp[blk];
    #pragma unroll
    for (int j = 0; j < 8; ++j) {
      float f = (float)v[j];
      s += f * att_src[hd * OUT_F + blk * 8 + j];
      d += f * att_dst[hd * OUT_F + blk * 8 + j];
    }
  }
  asrc[idx] = s; adst[idx] = d;
}

__global__ void k_edge_max(const int* __restrict__ ei, const float* __restrict__ asrc,
                           const float* __restrict__ adst, unsigned* __restrict__ amax,
                           int E_, int n) {
  int e = blockIdx.x * blockDim.x + threadIdx.x;
  if (e >= E_ + n) return;
  int s, d; edge_nodes(ei, e, E_, s, d);
  #pragma unroll
  for (int hd = 0; hd < HEADS; ++hd) {
    float al = asrc[s * HEADS + hd] + adst[d * HEADS + hd];
    al = (al >= 0.f) ? al : NEG_SLOPE * al;
    atomicMax(&amax[d * HEADS + hd], enc_f32(al));
  }
}

__global__ void k_edge_sum(const int* __restrict__ ei, const float* __restrict__ asrc,
                           const float* __restrict__ adst, const unsigned* __restrict__ amax,
                           float* __restrict__ denom, int E_, int n) {
  int e = blockIdx.x * blockDim.x + threadIdx.x;
  if (e >= E_ + n) return;
  int s, d; edge_nodes(ei, e, E_, s, d);
  #pragma unroll
  for (int hd = 0; hd < HEADS; ++hd) {
    float al = asrc[s * HEADS + hd] + adst[d * HEADS + hd];
    al = (al >= 0.f) ? al : NEG_SLOPE * al;
    atomicAdd(&denom[d * HEADS + hd], expf(al - dec_f32(amax[d * HEADS + hd])));
  }
}

// One wave per edge; lane owns 4 contiguous columns (cols lane*4..lane*4+3,
// head = lane>>3) -> h[src] gather is one contiguous 256B transaction per edge.
__global__ void k_scatter(const int* __restrict__ ei, const _Float16* __restrict__ h,
                          const float* __restrict__ asrc, const float* __restrict__ adst,
                          const unsigned* __restrict__ amax, const float* __restrict__ denom,
                          float* __restrict__ out, int E_, int n) {
  int gid  = blockIdx.x * blockDim.x + threadIdx.x;
  int e    = gid >> 5;
  int lane = gid & 31;
  if (e >= E_ + n) return;
  int s, d; edge_nodes(ei, e, E_, s, d);
  int hd = lane >> 3;   // 8 lanes per head (32 cols / 4 per lane)

  float al = asrc[s * HEADS + hd] + adst[d * HEADS + hd];
  al = (al >= 0.f) ? al : NEG_SLOPE * al;
  float w = expf(al - dec_f32(amax[d * HEADS + hd])) /
            (denom[d * HEADS + hd] + 1e-16f);

  v4h hv = ((const v4h*)(h + (size_t)s * IN_F))[lane];   // 8B load
  float* op = out + (size_t)d * IN_F + lane * 4;
  atomicAdd(op + 0, w * (float)hv[0]);
  atomicAdd(op + 1, w * (float)hv[1]);
  atomicAdd(op + 2, w * (float)hv[2]);
  atomicAdd(op + 3, w * (float)hv[3]);
}

extern "C" void kernel_launch(void* const* d_in, const int* in_sizes, int n_in,
                              void* d_out, int out_size, void* d_ws, size_t ws_size,
                              hipStream_t stream) {
  const float* x       = (const float*)d_in[0];
  const int*   ei      = (const int*)  d_in[1];   // [2,E] flat: src row then dst row
  const float* W       = (const float*)d_in[2];
  const float* att_src = (const float*)d_in[3];
  const float* att_dst = (const float*)d_in[4];
  const float* bias    = (const float*)d_in[5];
  float* out = (float*)d_out;

  int n  = in_sizes[0] / IN_F;
  int E_ = in_sizes[1] / 2;
  int Et = E_ + n;

  // workspace carve-out (256B aligned slabs)
  char* ws = (char*)d_ws;
  size_t off = 0;
  auto alloc = [&](size_t bytes) -> void* {
    void* p = ws + off;
    off += (bytes + 255) & ~(size_t)255;
    return p;
  };
  _Float16* Wp   = (_Float16*)alloc((size_t)WP_HALVES * sizeof(_Float16));  // 32 KB
  _Float16* h    = (_Float16*)alloc((size_t)n * IN_F * sizeof(_Float16));   // 25.6 MB
  float*    asrc = (float*)   alloc((size_t)n * HEADS * sizeof(float));
  float*    adst = (float*)   alloc((size_t)n * HEADS * sizeof(float));
  unsigned* amax = (unsigned*)alloc((size_t)n * HEADS * sizeof(unsigned));
  float*    den  = (float*)   alloc((size_t)n * HEADS * sizeof(float));

  int ntiles = (n + 15) / 16;

  k_init    <<<(n * IN_F + 255) / 256, 256, 0, stream>>>(out, amax, den, bias, n);
  k_pack_w  <<<(WP_HALVES + 255) / 256, 256, 0, stream>>>(W, Wp);
  k_gemm    <<<(ntiles + 3) / 4, 128, 0, stream>>>(x, Wp, h, n, ntiles);
  k_scores  <<<(n * HEADS + 255) / 256, 256, 0, stream>>>(h, att_src, att_dst, asrc, adst, n);
  k_edge_max<<<(Et + 255) / 256, 256, 0, stream>>>(ei, asrc, adst, amax, E_, n);
  k_edge_sum<<<(Et + 255) / 256, 256, 0, stream>>>(ei, asrc, adst, amax, den, E_, n);
  k_scatter <<<((Et * 32) + 255) / 256, 256, 0, stream>>>(ei, h, asrc, adst, amax, den, out, E_, n);
}
